// KOAMultiHeadAttention_59236188947239
// MI455X (gfx1250) — matrix operand order repaired and checked by reading, not compiled, not run on hardware
//
#include <hip/hip_runtime.h>

// ---------------------------------------------------------------------------
// Performer (FAVOR+) causal linear attention for MI455X (gfx1250, wave32).
// Matmuls on v_wmma_f32_16x16x32_bf16; bf16 tile staging via the Tensor Data
// Mover (tensor_load_to_lds + s_wait_tensorcnt); intermediates bf16 (fit L2).
// ---------------------------------------------------------------------------

typedef __attribute__((ext_vector_type(16))) __bf16 v16bf;
typedef __attribute__((ext_vector_type(8)))  float  v8f;
typedef __attribute__((ext_vector_type(4)))  unsigned int v4u;
typedef __attribute__((ext_vector_type(8)))  int    v8i;
typedef __attribute__((ext_vector_type(4)))  int    v4i;
typedef unsigned int u32;

#define Tt     2048
#define Hh     16
#define DKk    64
#define Mm     128
#define NCHUNK 16

// ---- Tensor Data Mover: 2-D bf16 tile (global -> LDS), D# per ISA 8.3/8.4 --
// Elements are 2 bytes. tile_d0 elems/row, tile_d1 rows, row stride stride_d0
// elems; LDS receives the tile packed contiguously.
__device__ __forceinline__ void tdm_load_2d_bf16(void* lds, const void* gptr,
                                                 u32 tile_d0, u32 tile_d1,
                                                 u32 stride_d0) {
  u32 lds_addr = (u32)(size_t)lds;                    // low 32 bits = LDS offset
  unsigned long long ga = (unsigned long long)(size_t)gptr;
  v4u g0;
  g0[0] = 1u;                                         // count=1, user D#
  g0[1] = lds_addr;                                   // lds_addr
  g0[2] = (u32)ga;                                    // global_addr[31:0]
  g0[3] = (u32)((ga >> 32) & 0x1FFFFFFu) | (2u << 30); // addr[56:32] | type=2
  v8i g1;
  g1[0] = (int)(1u << 16);                            // data_size=2B, no mcast
  g1[1] = (int)((tile_d0 & 0xFFFFu) << 16);           // tensor_dim0[15:0]
  g1[2] = (int)(((tile_d0 >> 16) & 0xFFFFu) |         // tensor_dim0[31:16]
                ((tile_d1 & 0xFFFFu) << 16));         // tensor_dim1[15:0]
  g1[3] = (int)(((tile_d1 >> 16) & 0xFFFFu) |         // tensor_dim1[31:16]
                ((tile_d0 & 0xFFFFu) << 16));         // tile_dim0
  g1[4] = (int)(tile_d1 & 0xFFFFu);                   // tile_dim1 (tile_dim2=0)
  g1[5] = (int)stride_d0;                             // tensor_dim0_stride[31:0]
  g1[6] = 0;
  g1[7] = 0;
  v4i z4 = {0, 0, 0, 0};
#if __has_include(<hip/amd_detail/amd_gfx1250_TDM.h>)
  v8i z8 = {0, 0, 0, 0, 0, 0, 0, 0};
  __builtin_amdgcn_tensor_load_to_lds(g0, g1, z4, z4, z8, 0);
#else
  __builtin_amdgcn_tensor_load_to_lds(g0, g1, z4, z4, 0);
#endif
}
__device__ __forceinline__ void tdm_wait() {
  __builtin_amdgcn_s_wait_tensorcnt(0);
}

// ---- WMMA fragment helpers (contiguous k = 16*(lane/16) + e in both A and B, wave32) ----------
__device__ __forceinline__ v16bf frag_a(const __bf16* p, int rs, int ks, int r0, int k0) {
  const int lane = threadIdx.x & 31;
  const int r = r0 + (lane & 15);
  const int hf = lane >> 4;
  v16bf f;
#pragma unroll
  for (int e = 0; e < 16; ++e) {
    int k = k0 + ((hf << 4) | e);
    f[e] = p[r * rs + k * ks];
  }
  return f;
}
__device__ __forceinline__ v16bf frag_b(const __bf16* p, int ns, int ks, int n0, int k0) {
  const int lane = threadIdx.x & 31;
  const int n = n0 + (lane & 15);
  const int hf = lane >> 4;
  v16bf f;
#pragma unroll
  for (int e = 0; e < 16; ++e) {
    int k = k0 + ((hf << 4) | e);
    f[e] = p[n * ns + k * ks];
  }
  return f;
}
__device__ __forceinline__ v8f wmma_bf16(v16bf a, v16bf b, v8f c) {
  return __builtin_amdgcn_wmma_f32_16x16x32_bf16(false, a, false, b, (short)0, c,
                                                 false, false);
}

// ---------------------------------------------------------------------------
// Kernel 1/6: C[8192,1024] = A[8192,1024] * W[1024,1024]^T + bias
// (fp32 sources -> bf16 conversion in staging, so VALU staging path here.)
// ---------------------------------------------------------------------------
template <typename TIN, typename TOUT>
__global__ __launch_bounds__(256) void gemm_xt_kernel(const TIN* __restrict__ A,
                                                      const float* __restrict__ W,
                                                      const float* __restrict__ bias,
                                                      TOUT* __restrict__ out) {
  __shared__ __bf16 sA[128 * 32];
  __shared__ __bf16 sB[128 * 32];
  const int tid = threadIdx.x;
  const int wid = tid >> 5;
  const int lane = tid & 31;
  const int l15 = lane & 15, hf = lane >> 4;
  const int m0 = blockIdx.y * 128;
  const int n0 = blockIdx.x * 128;
  v8f acc[8];
#pragma unroll
  for (int c = 0; c < 8; ++c)
#pragma unroll
    for (int i = 0; i < 8; ++i) acc[c][i] = 0.f;

  for (int kt = 0; kt < 1024; kt += 32) {
#pragma unroll
    for (int j = 0; j < 16; ++j) {
      int idx = tid + j * 256;  // 4096 elems each panel
      int r = idx >> 5, cc = idx & 31;
      sA[idx] = (__bf16)(float)A[(size_t)(m0 + r) * 1024 + kt + cc];
      sB[idx] = (__bf16)W[(size_t)(n0 + r) * 1024 + kt + cc];
    }
    __syncthreads();
    v16bf af = frag_a(sA, 32, 1, wid * 16, 0);
#pragma unroll
    for (int c = 0; c < 8; ++c) {
      v16bf bf = frag_b(sB, 32, 1, c * 16, 0);
      acc[c] = wmma_bf16(af, bf, acc[c]);
    }
    __syncthreads();
  }
#pragma unroll
  for (int c = 0; c < 8; ++c)
#pragma unroll
    for (int i = 0; i < 8; ++i) {
      int row = m0 + wid * 16 + i + hf * 8;
      int col = n0 + c * 16 + l15;
      out[(size_t)row * 1024 + col] = (TOUT)(acc[c][i] + bias[col]);
    }
}

// ---------------------------------------------------------------------------
// Kernel 2: phi features.  X tile staged by TDM (64x128 tile, stride 1024).
// ---------------------------------------------------------------------------
__global__ __launch_bounds__(256) void phi_kernel(const __bf16* __restrict__ X,
                                                  const float* __restrict__ proj,
                                                  __bf16* __restrict__ feat) {
  __shared__ __bf16 sX[128 * 64];
  __shared__ __bf16 sP[128 * 64];
  __shared__ float ssq[128];
  const int tid = threadIdx.x, wid = tid >> 5, lane = tid & 31;
  const int l15 = lane & 15, hf = lane >> 4;
  const int bh = blockIdx.y;          // b*H + h
  const int b = bh >> 4, h = bh & 15;
  const int t0 = blockIdx.x * 128;
  if (wid == 0) {
    tdm_load_2d_bf16(sX, X + (size_t)(b * Tt + t0) * 1024 + h * 64,
                     /*d0=*/64, /*d1=*/128, /*stride=*/1024);
  }
#pragma unroll
  for (int j = 0; j < 32; ++j) {
    int idx = tid + j * 256;          // proj needs f32->bf16: VALU staging
    sP[idx] = (__bf16)proj[idx];      // proj is [M=128, dk=64] row-major
  }
  if (wid == 0) tdm_wait();
  __syncthreads();
  if (tid < 128) {
    float s = 0.f;
    for (int d = 0; d < 64; ++d) {
      float x = (float)sX[tid * 64 + d];
      s += x * x;
    }
    ssq[tid] = 0.0625f * s;           // 0.5 * (dk^-0.5) * |x|^2
  }
  __syncthreads();
  v8f acc[8];
#pragma unroll
  for (int c = 0; c < 8; ++c)
#pragma unroll
    for (int i = 0; i < 8; ++i) acc[c][i] = 0.f;
#pragma unroll
  for (int k0 = 0; k0 < 64; k0 += 32) {
    v16bf af = frag_a(sX, 64, 1, wid * 16, k0);
#pragma unroll
    for (int c = 0; c < 8; ++c) {
      v16bf bf = frag_b(sP, 64, 1, c * 16, k0);
      acc[c] = wmma_bf16(af, bf, acc[c]);
    }
  }
  const float S1 = 0.35355339059327373f;    // dk^-0.25
  const float INVSM = 0.08838834764831845f; // 1/sqrt(M)
#pragma unroll
  for (int c = 0; c < 8; ++c)
#pragma unroll
    for (int i = 0; i < 8; ++i) {
      int tr = wid * 16 + i + hf * 8;
      int mc = c * 16 + l15;
      float v = __expf(acc[c][i] * S1 - ssq[tr]) * INVSM + 1e-6f;
      feat[((size_t)bh * Tt + t0 + tr) * 128 + mc] = (__bf16)v;
    }
}

// ---------------------------------------------------------------------------
// Kernel 3: per-chunk kv[m,d] = sum_k kf[k,m]*v[k,d],  zc[m] = sum_k kf[k,m]
// kf chunk (contiguous 32KB) and v tile staged by TDM.
// ---------------------------------------------------------------------------
__global__ __launch_bounds__(256) void chunk_kv_kernel(const __bf16* __restrict__ kf,
                                                       const __bf16* __restrict__ Vb,
                                                       float* __restrict__ kvb,
                                                       float* __restrict__ zcb) {
  __shared__ __bf16 skf[128 * 128];
  __shared__ __bf16 svc[128 * 64];
  const int tid = threadIdx.x, wid = tid >> 5, lane = tid & 31;
  const int l15 = lane & 15, hf = lane >> 4;
  const int blk = blockIdx.x;
  const int bh = blk >> 4, n = blk & 15;
  const int b = bh >> 4, h = bh & 15;
  const size_t krow = (size_t)bh * Tt + n * 128;
  const size_t vrow = (size_t)b * Tt + n * 128;
  if (wid == 0) {
    tdm_load_2d_bf16(skf, kf + krow * 128, 128, 128, 128);
    tdm_load_2d_bf16(svc, Vb + vrow * 1024 + h * 64, 64, 128, 1024);
    tdm_wait();
  }
  __syncthreads();
  if (tid < 128) {
    float s = 0.f;
    for (int k = 0; k < 128; ++k) s += (float)skf[k * 128 + tid];
    zcb[(size_t)blk * 128 + tid] = s;
  }
  v8f acc[4];
#pragma unroll
  for (int c = 0; c < 4; ++c)
#pragma unroll
    for (int i = 0; i < 8; ++i) acc[c][i] = 0.f;
  for (int k0 = 0; k0 < 128; k0 += 32) {
    v16bf af = frag_a(skf, 1, 128, wid * 16, k0);  // A[m,k] = kf[k,m]
#pragma unroll
    for (int c = 0; c < 4; ++c) {
      v16bf bf = frag_b(svc, 1, 64, c * 16, k0);   // B[k,d] = v[k,d]
      acc[c] = wmma_bf16(af, bf, acc[c]);
    }
  }
#pragma unroll
  for (int c = 0; c < 4; ++c)
#pragma unroll
    for (int i = 0; i < 8; ++i) {
      int m = wid * 16 + i + hf * 8;
      int d = c * 16 + l15;
      kvb[(size_t)blk * 8192 + m * 64 + d] = acc[c][i];
    }
}

// ---------------------------------------------------------------------------
// Kernel 4: exclusive prefix sums over chunks (in place) per (b,h).
// ---------------------------------------------------------------------------
__global__ __launch_bounds__(256) void prefix_kernel(float* __restrict__ kvb,
                                                     float* __restrict__ zcb) {
  const int bh = blockIdx.x, tid = threadIdx.x;
  float acc[32];
#pragma unroll
  for (int j = 0; j < 32; ++j) acc[j] = 0.f;
  for (int n = 0; n < NCHUNK; ++n) {
    size_t base = ((size_t)bh * NCHUNK + n) * 8192;
#pragma unroll
    for (int j = 0; j < 32; ++j) {
      size_t idx = base + tid + j * 256;
      float v = kvb[idx];
      kvb[idx] = acc[j];
      acc[j] += v;
    }
  }
  if (tid < 128) {
    float z = 0.f;
    for (int n = 0; n < NCHUNK; ++n) {
      size_t idx = ((size_t)bh * NCHUNK + n) * 128 + tid;
      float v = zcb[idx];
      zcb[idx] = z;
      z += v;
    }
  }
}

// ---------------------------------------------------------------------------
// Kernel 5: per-chunk causal attention.
//   A = tril(qf kf^T); num = qf S_prev + A v; den = qf z_prev + rowsum(A) + eps
// qf/kf chunks + v tile staged by TDM; S_prev converts f32->bf16 on VALU.
// ---------------------------------------------------------------------------
__global__ __launch_bounds__(256) void chunk_attn_kernel(const __bf16* __restrict__ qf,
                                                         const __bf16* __restrict__ kf,
                                                         const __bf16* __restrict__ Vb,
                                                         const float* __restrict__ Sp,
                                                         const float* __restrict__ zp,
                                                         __bf16* __restrict__ attn) {
  __shared__ __bf16 sqf[128 * 128];  // qf chunk  [t][m]
  __shared__ __bf16 skA[128 * 128];  // kf chunk, later masked A (bf16)
  __shared__ __bf16 svc[128 * 64];   // v chunk   [t][d]
  __shared__ __bf16 sS[128 * 64];    // S_prev    [m][d] (bf16)
  __shared__ float szp[128];
  __shared__ float sden[128];
  const int tid = threadIdx.x, wid = tid >> 5, lane = tid & 31;
  const int l15 = lane & 15, hf = lane >> 4;
  const int blk = blockIdx.x;
  const int bh = blk >> 4, n = blk & 15;
  const int b = bh >> 4, h = bh & 15;
  const size_t frow = (size_t)bh * Tt + n * 128;
  const size_t vrow = (size_t)b * Tt + n * 128;
  if (wid == 0) {
    tdm_load_2d_bf16(sqf, qf + frow * 128, 128, 128, 128);
    tdm_load_2d_bf16(skA, kf + frow * 128, 128, 128, 128);
    tdm_load_2d_bf16(svc, Vb + vrow * 1024 + h * 64, 64, 128, 1024);
  }
#pragma unroll
  for (int j = 0; j < 32; ++j) {
    int idx = tid + j * 256;
    sS[idx] = (__bf16)Sp[(size_t)blk * 8192 + idx];   // f32 -> bf16 on VALU
  }
  if (tid < 128) {
    szp[tid] = zp[(size_t)blk * 128 + tid];
    sden[tid] = 0.f;
  }
  if (wid == 0) tdm_wait();
  __syncthreads();

  // ---- A = qf @ kf^T, masked causal, per-row sums via LDS atomics ----
  v8f a8[8];
#pragma unroll
  for (int c = 0; c < 8; ++c)
#pragma unroll
    for (int i = 0; i < 8; ++i) a8[c][i] = 0.f;
  for (int k0 = 0; k0 < 128; k0 += 32) {
    v16bf af = frag_a(sqf, 128, 1, wid * 16, k0);
#pragma unroll
    for (int c = 0; c < 8; ++c) {
      v16bf bf = frag_b(skA, 128, 1, c * 16, k0);  // B[m][t2] = kf[t2][m]
      a8[c] = wmma_bf16(af, bf, a8[c]);
    }
  }
  float rs[8];
#pragma unroll
  for (int i = 0; i < 8; ++i) rs[i] = 0.f;
#pragma unroll
  for (int c = 0; c < 8; ++c)
#pragma unroll
    for (int i = 0; i < 8; ++i) {
      int t1 = wid * 16 + i + hf * 8;
      int t2 = c * 16 + l15;
      float v = (t2 <= t1) ? a8[c][i] : 0.f;
      a8[c][i] = v;
      rs[i] += v;
    }
#pragma unroll
  for (int i = 0; i < 8; ++i) atomicAdd(&sden[wid * 16 + i + hf * 8], rs[i]);
  __syncthreads();  // all waves done reading kf
#pragma unroll
  for (int c = 0; c < 8; ++c)
#pragma unroll
    for (int i = 0; i < 8; ++i) {
      int t1 = wid * 16 + i + hf * 8;
      int t2 = c * 16 + l15;
      skA[t1 * 128 + t2] = (__bf16)a8[c][i];  // reuse kf buffer for A
    }
  __syncthreads();

  // ---- total denominator: intra (atomics) + inter (qf . z_prev) + eps ----
  if (tid < 128) {
    float s = 0.f;
    for (int m = 0; m < 128; ++m) s += (float)sqf[tid * 128 + m] * szp[m];
    sden[tid] = sden[tid] + s + 1e-6f;
  }

  // ---- numerator: qf @ S_prev  +  A @ v  (shared accumulators) ----
  v8f acc[4];
#pragma unroll
  for (int c = 0; c < 4; ++c)
#pragma unroll
    for (int i = 0; i < 8; ++i) acc[c][i] = 0.f;
  for (int k0 = 0; k0 < 128; k0 += 32) {
    v16bf af = frag_a(sqf, 128, 1, wid * 16, k0);
#pragma unroll
    for (int c = 0; c < 4; ++c) {
      v16bf bf = frag_b(sS, 1, 64, c * 16, k0);   // B[m][d] = S_prev[m][d]
      acc[c] = wmma_bf16(af, bf, acc[c]);
    }
  }
  for (int k0 = 0; k0 < 128; k0 += 32) {
    v16bf af = frag_a(skA, 128, 1, wid * 16, k0); // masked A
#pragma unroll
    for (int c = 0; c < 4; ++c) {
      v16bf bf = frag_b(svc, 1, 64, c * 16, k0);  // B[k][d] = v[k][d]
      acc[c] = wmma_bf16(af, bf, acc[c]);
    }
  }
  __syncthreads();  // sden ready
#pragma unroll
  for (int c = 0; c < 4; ++c)
#pragma unroll
    for (int i = 0; i < 8; ++i) {
      int t1 = wid * 16 + i + hf * 8;
      int d = c * 16 + l15;
      float o = acc[c][i] / sden[t1];
      attn[(size_t)(b * Tt + n * 128 + t1) * 1024 + h * 64 + d] = (__bf16)o;
    }
}

// ---------------------------------------------------------------------------
// kernel_launch: x->(Q,K,V)->phi->chunk kv->prefix->chunk attn->Wo GEMM
// Workspace (bf16 intermediates, ~112.5 MB; kv scratch lives in d_out):
//   [0,16M)   Q (bf16)  -> reused as attn
//   [16,32M)  K (bf16)
//   [32,48M)  V (bf16)
//   [48,80M)  qf (bf16)
//   [80,112M) kf (bf16)
//   [112M,+512K) zc (f32)
// ---------------------------------------------------------------------------
extern "C" void kernel_launch(void* const* d_in, const int* in_sizes, int n_in,
                              void* d_out, int out_size, void* d_ws, size_t ws_size,
                              hipStream_t stream) {
  (void)in_sizes; (void)n_in; (void)out_size;
  const float* x    = (const float*)d_in[0];
  const float* Wq   = (const float*)d_in[1];
  const float* bq   = (const float*)d_in[2];
  const float* Wk   = (const float*)d_in[3];
  const float* bk   = (const float*)d_in[4];
  const float* Wv   = (const float*)d_in[5];
  const float* bv   = (const float*)d_in[6];
  const float* Wo   = (const float*)d_in[7];
  const float* bo   = (const float*)d_in[8];
  const float* proj = (const float*)d_in[9];
  float* out = (float*)d_out;

  char* ws = (char*)d_ws;
  if (ws_size < ((size_t)113 << 20)) return;
  __bf16* Qb  = (__bf16*)(ws + ((size_t)0 << 20));
  __bf16* Kb  = (__bf16*)(ws + ((size_t)16 << 20));
  __bf16* Vb  = (__bf16*)(ws + ((size_t)32 << 20));
  __bf16* qfb = (__bf16*)(ws + ((size_t)48 << 20));
  __bf16* kfb = (__bf16*)(ws + ((size_t)80 << 20));
  float*  zcb = (float*)(ws + ((size_t)112 << 20));
  float*  kvb = out;            // 8192*1024 f32 == 1024 chunks * 128*64 f32
  __bf16* attnb = Qb;           // Q dead after phi; reuse as attention output

  dim3 blk(256);
  dim3 g_gemm(8, 64);           // 1024/128 cols x 8192/128 rows
  gemm_xt_kernel<float, __bf16><<<g_gemm, blk, 0, stream>>>(x, Wq, bq, Qb);
  gemm_xt_kernel<float, __bf16><<<g_gemm, blk, 0, stream>>>(x, Wk, bk, Kb);
  gemm_xt_kernel<float, __bf16><<<g_gemm, blk, 0, stream>>>(x, Wv, bv, Vb);

  dim3 g_phi(16, 64);           // T/128 x B*H
  phi_kernel<<<g_phi, blk, 0, stream>>>(Qb, proj, qfb);
  phi_kernel<<<g_phi, blk, 0, stream>>>(Kb, proj, kfb);

  chunk_kv_kernel<<<dim3(1024), blk, 0, stream>>>(kfb, Vb, kvb, zcb);
  prefix_kernel<<<dim3(64), blk, 0, stream>>>(kvb, zcb);
  chunk_attn_kernel<<<dim3(1024), blk, 0, stream>>>(qfb, kfb, Vb, kvb, zcb, attnb);

  gemm_xt_kernel<__bf16, float><<<g_gemm, blk, 0, stream>>>(attnb, Wo, bo, out);
}
